// SiglipEncoderLayer_45921790329024
// MI455X (gfx1250) — matrix operand order repaired and checked by reading, not compiled, not run on hardware
//
#include <hip/hip_runtime.h>
#include <hip/hip_bf16.h>

// ---------------------------------------------------------------------------
// SigLIP encoder layer forward for MI455X (gfx1250).
// bf16 WMMA GEMMs + flash attention; async global->LDS tile staging (ASYNCcnt).
// ---------------------------------------------------------------------------

typedef __bf16 bf16_t;
typedef __attribute__((ext_vector_type(16))) __bf16 v16bf;
typedef __attribute__((ext_vector_type(8)))  float  v8f;

static constexpr int   Bb     = 8;
static constexpr int   Nn     = 729;
static constexpr int   Dd     = 1152;
static constexpr int   Hh     = 12;
static constexpr int   HDd    = 96;
static constexpr int   HIDn   = 4304;
static constexpr int   HIDP   = 4320;            // padded to /32 for K of fc2
static constexpr int   Mm     = Bb * Nn;         // 5832 rows
static constexpr float QSCALE = 0.102062072615966f; // 96^-0.5

union Frag { v16bf v; uint4 u[2]; };

__device__ __forceinline__ v8f wmma_bf16(const Frag& a, const Frag& b, v8f c) {
  return __builtin_amdgcn_wmma_f32_16x16x32_bf16(false, a.v, false, b.v,
                                                 (short)0, c, false, false);
}

// Async global->LDS 16B copy (per-lane), tracked by ASYNCcnt.
__device__ __forceinline__ void async_ld_b128(unsigned lds_byte_addr, const void* g) {
  asm volatile("global_load_async_to_lds_b128 %0, %1, off"
               :: "v"(lds_byte_addr), "v"((unsigned long long)(uintptr_t)g)
               : "memory");
}
__device__ __forceinline__ void wait_async0() {
  asm volatile("s_wait_asynccnt 0x0" ::: "memory");
}
__device__ __forceinline__ void wait_async4() {
  asm volatile("s_wait_asynccnt 0x4" ::: "memory");
}
__device__ __forceinline__ unsigned lds_addr(const void* p) {
  return (unsigned)(uintptr_t)p;   // LDS aperture: byte offset in low 32 bits
}

// ---------------------------------------------------------------------------
// GEMM: C[M, ldc] = (A[M,K] @ W[Nout,K]^T + bias[Nout]) * scale
// A, W bf16 row-major; C fp32 or bf16 (OUT_BF16). K % 32 == 0.
// Block tile 128x128, BK=32, 8 waves -> each wave 64x32 (8 wmma / k-step).
// Double-buffered async global->LDS staging.
// ---------------------------------------------------------------------------
template<bool OUT_BF16>
__global__ __launch_bounds__(256)
void gemm_bf16_nt(const bf16_t* __restrict__ A, const bf16_t* __restrict__ W,
                  const float* __restrict__ bias, void* __restrict__ C,
                  int Mrows, int K, int Nout, int ldc, float scale) {
  constexpr int BM = 128, BN = 128, BK = 32, LDK = 40; // LDK padded (bank spread)
  constexpr int TILE = BM * LDK;                       // elems per buffer
  __shared__ __align__(16) bf16_t sA[2 * TILE];
  __shared__ __align__(16) bf16_t sB[2 * TILE];

  const int tid  = threadIdx.x;
  const int wid  = tid >> 5;
  const int lane = tid & 31;
  const int r    = lane & 15;
  const int hh   = lane >> 4;
  const int bm   = blockIdx.x * BM;
  const int bn   = blockIdx.y * BN;
  const int waveM = (wid & 1) * 64;   // 2 waves along M
  const int waveN = (wid >> 1) * 32;  // 4 waves along N

  // Staging map: thread covers chunks (row1, col8) and (row1+64, col8).
  const int row1 = tid >> 2;          // 0..63
  const int col8 = (tid & 3) * 8;
  int gA1 = bm + row1;       if (gA1 >= Mrows) gA1 = Mrows - 1;  // clamp (safe)
  int gA2 = bm + row1 + 64;  if (gA2 >= Mrows) gA2 = Mrows - 1;
  int gB1 = bn + row1;       if (gB1 >= Nout)  gB1 = Nout - 1;
  int gB2 = bn + row1 + 64;  if (gB2 >= Nout)  gB2 = Nout - 1;
  const unsigned lA1 = lds_addr(sA + row1 * LDK + col8);
  const unsigned lA2 = lA1 + 64 * LDK * 2;
  const unsigned lB1 = lds_addr(sB + row1 * LDK + col8);
  const unsigned lB2 = lB1 + 64 * LDK * 2;

  auto issue = [&](int kt, int sel) {
    const size_t k0 = (size_t)kt * BK + col8;
    const unsigned o = (unsigned)sel * (unsigned)(TILE * 2);   // bytes
    async_ld_b128(lA1 + o, A + (size_t)gA1 * K + k0);
    async_ld_b128(lA2 + o, A + (size_t)gA2 * K + k0);
    async_ld_b128(lB1 + o, W + (size_t)gB1 * K + k0);
    async_ld_b128(lB2 + o, W + (size_t)gB2 * K + k0);
  };

  v8f acc[4][2] = {};
  const int nk = K / BK;
  issue(0, 0);
  for (int kt = 0; kt < nk; ++kt) {
    const int sel = kt & 1;
    if (kt + 1 < nk) {            // prefetch next tile into the other buffer
      issue(kt + 1, sel ^ 1);
      wait_async4();              // oldest 4 (this tile) have landed in LDS
    } else {
      wait_async0();
    }
    __syncthreads();

    const bf16_t* dA = sA + sel * TILE;
    const bf16_t* dB = sB + sel * TILE;

    // A-fragment: lane row = r; K chunks hh*8 and 16+hh*8.
    Frag fa[4];
#pragma unroll
    for (int i = 0; i < 4; ++i) {
      const bf16_t* p = dA + (waveM + i * 16 + r) * LDK + hh * 8;
      fa[i].u[0] = *(const uint4*)p;
      fa[i].u[1] = *(const uint4*)(p + 16);
    }
    // B-fragment: lane col = r; K chunks hh*8 and 16+hh*8.
    Frag fb[2];
#pragma unroll
    for (int j = 0; j < 2; ++j) {
      const bf16_t* p = dB + (waveN + j * 16 + r) * LDK + hh * 8;
      fb[j].u[0] = *(const uint4*)p;
      fb[j].u[1] = *(const uint4*)(p + 16);
    }
#pragma unroll
    for (int i = 0; i < 4; ++i)
#pragma unroll
      for (int j = 0; j < 2; ++j)
        acc[i][j] = wmma_bf16(fa[i], fb[j], acc[i][j]);
    __syncthreads();
  }

  // Epilogue. C layout: lane r = col, vgpr rr = row (+8 for upper lane half).
#pragma unroll
  for (int i = 0; i < 4; ++i) {
    const int rowBase = bm + waveM + i * 16 + hh * 8;
#pragma unroll
    for (int j = 0; j < 2; ++j) {
      const int col = bn + waveN + j * 16 + r;
      const float bia = (col < Nout) ? bias[col] : 0.0f;
      if (col < ldc) {
#pragma unroll
        for (int rr = 0; rr < 8; ++rr) {
          const int row = rowBase + rr;
          if (row < Mrows) {
            const float v = (acc[i][j][rr] + bia) * scale;
            if (OUT_BF16)
              ((bf16_t*)C)[(size_t)row * ldc + col] = (bf16_t)v;
            else
              ((float*)C)[(size_t)row * ldc + col] = v;
          }
        }
      }
    }
  }
}

// ---------------------------------------------------------------------------
// Flash-style attention. Q is pre-scaled by HD^-0.5. Per head the reference's
// plain-view split gives contiguous [729,96] Q/K/V slabs at
// base = b*N*D + h*N*HD. Output ctx[b, q, h*96 + d] bf16.
// Grid: (B*H, ceil(N/128)). 8 waves, each owns 16 query rows.
// ---------------------------------------------------------------------------
__global__ __launch_bounds__(256)
void attention_kernel(const bf16_t* __restrict__ Qb, const bf16_t* __restrict__ Kb,
                      const bf16_t* __restrict__ Vb, bf16_t* __restrict__ ctx) {
  constexpr int NKT = 64;                 // key tile
  constexpr int LDS_K = 104;              // 96 + pad (elems)
  constexpr int LDS_V = 72;               // 64 + pad
  constexpr int LDS_P = 72;               // 64 + pad
  __shared__ __align__(16) bf16_t sK[NKT * LDS_K];
  __shared__ __align__(16) bf16_t sVt[HDd * LDS_V];
  __shared__ __align__(16) bf16_t sP[8 * 16 * LDS_P];

  const int tid  = threadIdx.x;
  const int wid  = tid >> 5;
  const int lane = tid & 31;
  const int r    = lane & 15;
  const int hh   = lane >> 4;

  const int bh    = blockIdx.x;
  const int batch = bh / Hh;
  const int head  = bh % Hh;
  const int qt    = blockIdx.y * 128;
  const size_t hbase = (size_t)batch * ((size_t)Nn * Dd) + (size_t)head * (Nn * HDd);

  // Q fragments (3 K-steps of 32 over HD=96), kept in registers.
  Frag qf[3];
  {
    int qr = qt + wid * 16 + r;
    if (qr > Nn - 1) qr = Nn - 1;          // clamp (pad rows never stored)
    const bf16_t* qp = Qb + hbase + (size_t)qr * HDd;
#pragma unroll
    for (int kk = 0; kk < 3; ++kk) {
      qf[kk].u[0] = *(const uint4*)(qp + kk * 32 + hh * 8);
      qf[kk].u[1] = *(const uint4*)(qp + kk * 32 + 16 + hh * 8);
    }
  }

  float mrow[8], lrow[8];
#pragma unroll
  for (int rr = 0; rr < 8; ++rr) { mrow[rr] = -1e30f; lrow[rr] = 0.0f; }
  v8f cacc[6] = {};

  bf16_t* sPw = sP + wid * 16 * LDS_P;
  const int nkt = (Nn + NKT - 1) / NKT;    // 12

  for (int kt = 0; kt < nkt; ++kt) {
    // ---- Stage K tile via async global->LDS (clamped rows; masked later) ----
    for (int idx = tid; idx < NKT * 12; idx += 256) {   // 64 rows x 12 chunks
      const int kr = idx / 12, c8 = (idx % 12) * 8;
      int gk = kt * NKT + kr;
      if (gk > Nn - 1) gk = Nn - 1;
      async_ld_b128(lds_addr(sK + kr * LDS_K + c8),
                    Kb + hbase + (size_t)gk * HDd + c8);
    }
    // ---- Stage V transposed (scalar path; zero-fill pad rows) ----
    for (int idx = tid; idx < NKT * HDd; idx += 256) {
      const int kr = idx / HDd, d = idx % HDd;
      const int gk = kt * NKT + kr;
      bf16_t v = (bf16_t)0.0f;
      if (gk < Nn) v = Vb[hbase + (size_t)gk * HDd + d];
      sVt[d * LDS_V + kr] = v;
    }
    wait_async0();
    __syncthreads();

    // ---- S = Q K^T  (4 key sub-tiles of 16) ----
    v8f s[4] = {};
#pragma unroll
    for (int j = 0; j < 4; ++j)
#pragma unroll
      for (int kk = 0; kk < 3; ++kk) {
        Frag fb;
        const bf16_t* p = sK + (j * 16 + r) * LDS_K + kk * 32 + hh * 8;
        fb.u[0] = *(const uint4*)p;
        fb.u[1] = *(const uint4*)(p + 16);
        s[j] = wmma_bf16(qf[kk], fb, s[j]);
      }

    // ---- mask pad keys, row max ----
    float tmax[8];
#pragma unroll
    for (int rr = 0; rr < 8; ++rr) tmax[rr] = -1e30f;
#pragma unroll
    for (int j = 0; j < 4; ++j) {
      const bool oob = (kt * NKT + j * 16 + r) >= Nn;
#pragma unroll
      for (int rr = 0; rr < 8; ++rr) {
        float v = oob ? -1e30f : s[j][rr];
        s[j][rr] = v;
        tmax[rr] = fmaxf(tmax[rr], v);
      }
    }
#pragma unroll
    for (int msk = 1; msk <= 8; msk <<= 1)
#pragma unroll
      for (int rr = 0; rr < 8; ++rr)
        tmax[rr] = fmaxf(tmax[rr], __shfl_xor(tmax[rr], msk, 32));

    float scl[8], psum[8];
#pragma unroll
    for (int rr = 0; rr < 8; ++rr) {
      const float mnew = fmaxf(mrow[rr], tmax[rr]);
      scl[rr]  = __expf(mrow[rr] - mnew);
      mrow[rr] = mnew;
      lrow[rr] *= scl[rr];
      psum[rr] = 0.0f;
    }

    // ---- P = exp(S - m), stage per-wave into LDS ----
#pragma unroll
    for (int j = 0; j < 4; ++j)
#pragma unroll
      for (int rr = 0; rr < 8; ++rr) {
        const float p = __expf(s[j][rr] - mrow[rr]);
        psum[rr] += p;
        sPw[(hh * 8 + rr) * LDS_P + j * 16 + r] = (bf16_t)p;
      }
#pragma unroll
    for (int msk = 1; msk <= 8; msk <<= 1)
#pragma unroll
      for (int rr = 0; rr < 8; ++rr)
        psum[rr] += __shfl_xor(psum[rr], msk, 32);
#pragma unroll
    for (int rr = 0; rr < 8; ++rr) lrow[rr] += psum[rr];

    // rescale running context
#pragma unroll
    for (int jd = 0; jd < 6; ++jd)
#pragma unroll
      for (int rr = 0; rr < 8; ++rr) cacc[jd][rr] *= scl[rr];

    asm volatile("" ::: "memory");   // keep P stores before P loads (same wave)

    // ---- ctx += P @ V  (A = P rows, B = V^T rows) ----
#pragma unroll
    for (int kkp = 0; kkp < 2; ++kkp) {
      Frag fa;
      const bf16_t* pp = sPw + r * LDS_P + kkp * 32 + hh * 8;
      fa.u[0] = *(const uint4*)pp;
      fa.u[1] = *(const uint4*)(pp + 16);
#pragma unroll
      for (int jd = 0; jd < 6; ++jd) {
        Frag fb;
        const bf16_t* vp = sVt + (jd * 16 + r) * LDS_V + kkp * 32 + hh * 8;
        fb.u[0] = *(const uint4*)vp;
        fb.u[1] = *(const uint4*)(vp + 16);
        cacc[jd] = wmma_bf16(fa, fb, cacc[jd]);
      }
    }
    __syncthreads();
  }

  // ---- normalize + store: ctx[b, q, head*96 + d] ----
#pragma unroll
  for (int jd = 0; jd < 6; ++jd) {
    const int col = head * HDd + jd * 16 + r;
#pragma unroll
    for (int rr = 0; rr < 8; ++rr) {
      const int q = qt + wid * 16 + hh * 8 + rr;
      if (q < Nn) {
        const float v = cacc[jd][rr] / lrow[rr];
        ctx[((size_t)(batch * Nn + q)) * Dd + col] = (bf16_t)v;
      }
    }
  }
}

// ---------------------------------------------------------------------------
// LayerNorm(in)*g + b + res; optionally also emit gelu_tanh(result) as bf16.
// One row (D=1152) per block of 256.
// ---------------------------------------------------------------------------
template<bool WRITE_GELU>
__global__ __launch_bounds__(256)
void ln_residual(const float* __restrict__ in, const float* __restrict__ res,
                 const float* __restrict__ g, const float* __restrict__ be,
                 float* __restrict__ out_f, bf16_t* __restrict__ out_g) {
  __shared__ float red[18];
  const int tid  = threadIdx.x;
  const int wid  = tid >> 5;
  const int lane = tid & 31;
  const size_t row = blockIdx.x;
  const float* x = in + row * Dd;

  float s = 0.0f, s2 = 0.0f;
  for (int i = tid; i < Dd; i += 256) {
    const float v = x[i];
    s += v; s2 += v * v;
  }
#pragma unroll
  for (int msk = 1; msk < 32; msk <<= 1) {
    s  += __shfl_xor(s,  msk, 32);
    s2 += __shfl_xor(s2, msk, 32);
  }
  if (lane == 0) { red[wid] = s; red[8 + wid] = s2; }
  __syncthreads();
  if (tid == 0) {
    float ts = 0.0f, ts2 = 0.0f;
    for (int k = 0; k < 8; ++k) { ts += red[k]; ts2 += red[8 + k]; }
    const float mu  = ts / (float)Dd;
    const float var = ts2 / (float)Dd - mu * mu;
    red[16] = mu;
    red[17] = rsqrtf(var + 1e-6f);
  }
  __syncthreads();
  const float mu = red[16], rstd = red[17];
  for (int i = tid; i < Dd; i += 256) {
    const float v = (x[i] - mu) * rstd * g[i] + be[i] + res[row * Dd + i];
    out_f[row * Dd + i] = v;
    if (WRITE_GELU) {
      const float t   = 0.7978845608028654f * (v + 0.044715f * v * v * v);
      const float gel = 0.5f * v * (1.0f + tanhf(t));
      out_g[row * Dd + i] = (bf16_t)gel;
    }
  }
}

// ---------------------------------------------------------------------------
// fp32 -> bf16 converts
// ---------------------------------------------------------------------------
__global__ __launch_bounds__(256)
void cvt_f32_bf16(const float* __restrict__ in, bf16_t* __restrict__ out, long long n) {
  long long i = (long long)blockIdx.x * 256 + threadIdx.x;
  const long long stride = (long long)gridDim.x * 256;
  for (; i < n; i += stride) out[i] = (bf16_t)in[i];
}

// w_fc2 [D, HID] fp32 -> [D, HIDP] bf16 with zero padding in columns
__global__ __launch_bounds__(256)
void cvt_pad_fc2(const float* __restrict__ in, bf16_t* __restrict__ out) {
  long long i = (long long)blockIdx.x * 256 + threadIdx.x;
  const long long total = (long long)Dd * HIDP;
  const long long stride = (long long)gridDim.x * 256;
  for (; i < total; i += stride) {
    const int rr = (int)(i / HIDP);
    const int cc = (int)(i % HIDP);
    out[i] = (cc < HIDn) ? (bf16_t)in[(size_t)rr * HIDn + cc] : (bf16_t)0.0f;
  }
}

// ---------------------------------------------------------------------------
extern "C" void kernel_launch(void* const* d_in, const int* in_sizes, int n_in,
                              void* d_out, int out_size, void* d_ws, size_t ws_size,
                              hipStream_t stream) {
  const float* x    = (const float*)d_in[0];
  const float* wq   = (const float*)d_in[1];
  const float* bq   = (const float*)d_in[2];
  const float* wk   = (const float*)d_in[3];
  const float* bk   = (const float*)d_in[4];
  const float* wv   = (const float*)d_in[5];
  const float* bv   = (const float*)d_in[6];
  const float* wo   = (const float*)d_in[7];
  const float* bo   = (const float*)d_in[8];
  const float* ln1g = (const float*)d_in[9];
  const float* ln1b = (const float*)d_in[10];
  const float* wfc1 = (const float*)d_in[11];
  const float* bfc1 = (const float*)d_in[12];
  const float* wfc2 = (const float*)d_in[13];
  const float* bfc2 = (const float*)d_in[14];
  const float* ln2g = (const float*)d_in[15];
  const float* ln2b = (const float*)d_in[16];

  char* wptr = (char*)d_ws;
  auto carve = [&](size_t bytes) -> void* {
    void* p = (void*)wptr;
    wptr += (bytes + 255) & ~(size_t)255;
    return p;
  };

  const size_t MD = (size_t)Mm * Dd;
  bf16_t* xb    = (bf16_t*)carve(MD * 2);
  bf16_t* wqb   = (bf16_t*)carve((size_t)Dd * Dd * 2);
  bf16_t* wkb   = (bf16_t*)carve((size_t)Dd * Dd * 2);
  bf16_t* wvb   = (bf16_t*)carve((size_t)Dd * Dd * 2);
  bf16_t* wob   = (bf16_t*)carve((size_t)Dd * Dd * 2);
  bf16_t* wfc1b = (bf16_t*)carve((size_t)HIDn * Dd * 2);
  bf16_t* wfc2b = (bf16_t*)carve((size_t)Dd * HIDP * 2);
  bf16_t* Qb    = (bf16_t*)carve(MD * 2);
  bf16_t* Kb2   = (bf16_t*)carve(MD * 2);
  bf16_t* Vb2   = (bf16_t*)carve(MD * 2);
  bf16_t* ctxb  = (bf16_t*)carve(MD * 2);
  bf16_t* gbuf  = (bf16_t*)carve(MD * 2);
  bf16_t* m1    = (bf16_t*)carve((size_t)Mm * HIDP * 2);
  float*  attnf = (float*)carve(MD * 4);
  float*  hbuf  = (float*)carve(MD * 4);
  float*  m2    = (float*)carve(MD * 4);

  auto cvt = [&](const float* src, bf16_t* dst, long long n) {
    int blocks = (int)((n + 255) / 256);
    if (blocks > 4096) blocks = 4096;
    cvt_f32_bf16<<<blocks, 256, 0, stream>>>(src, dst, n);
  };
  cvt(x,    xb,    (long long)MD);
  cvt(wq,   wqb,   (long long)Dd * Dd);
  cvt(wk,   wkb,   (long long)Dd * Dd);
  cvt(wv,   wvb,   (long long)Dd * Dd);
  cvt(wo,   wob,   (long long)Dd * Dd);
  cvt(wfc1, wfc1b, (long long)HIDn * Dd);
  cvt_pad_fc2<<<4096, 256, 0, stream>>>(wfc2, wfc2b);

  const dim3 gD((Mm + 127) / 128, (Dd + 127) / 128);     // 46 x 9
  const dim3 gH((Mm + 127) / 128, (HIDP + 127) / 128);   // 46 x 34

  // Q (pre-scaled), K, V projections -> bf16
  gemm_bf16_nt<true><<<gD, 256, 0, stream>>>(xb, wqb, bq, Qb,  Mm, Dd, Dd, Dd, QSCALE);
  gemm_bf16_nt<true><<<gD, 256, 0, stream>>>(xb, wkb, bk, Kb2, Mm, Dd, Dd, Dd, 1.0f);
  gemm_bf16_nt<true><<<gD, 256, 0, stream>>>(xb, wvb, bv, Vb2, Mm, Dd, Dd, Dd, 1.0f);

  // Attention (flash-style, online softmax)
  attention_kernel<<<dim3(Bb * Hh, (Nn + 127) / 128), 256, 0, stream>>>(Qb, Kb2, Vb2, ctxb);

  // Output projection -> fp32
  gemm_bf16_nt<false><<<gD, 256, 0, stream>>>(ctxb, wob, bo, attnf, Mm, Dd, Dd, Dd, 1.0f);

  // h = LN(attn) + x ; gbuf = gelu(h) bf16
  ln_residual<true><<<Mm, 256, 0, stream>>>(attnf, x, ln1g, ln1b, hbuf, gbuf);

  // fc1: [M, 4304] (stored with padded stride 4320)
  gemm_bf16_nt<true><<<gH, 256, 0, stream>>>(gbuf, wfc1b, bfc1, m1, Mm, Dd, HIDn, HIDP, 1.0f);

  // fc2: K = 4320 (padded; pad weights are zero), output fp32
  gemm_bf16_nt<false><<<gD, 256, 0, stream>>>(m1, wfc2b, bfc2, m2, Mm, HIDP, Dd, Dd, 1.0f);

  // out = LN(m2) + h
  ln_residual<false><<<Mm, 256, 0, stream>>>(m2, hbuf, ln2g, ln2b, (float*)d_out, nullptr);
}
